// RNN_57896159150327
// MI455X (gfx1250) — matrix operand order repaired and checked by reading, not compiled, not run on hardware
//
#include <hip/hip_runtime.h>
#include <hip/hip_bf16.h>
#include <math.h>

// ---------------------------------------------------------------------------
// LSTM caption decoder for MI455X (gfx1250, wave32, WMMA + async-to-LDS).
// B=64, H=E=512, V=32000, T=50 steps, 4H=2048 gates.
//
//  * one-time: W_lin -> bf16 (L2-resident 32.8MB), W_cat=[W_ih|W_hh] bf16.
//  * per step: embed gather -> gates GEMM -> LSTM cell -> logits GEMM
//              -> argmax -> done reduce (stream-ordered; sequential argmax dep)
//  * GEMMs: A (64 rows, bf16) staged once per workgroup into padded LDS via
//    global_load_async_to_lds_b128 / s_wait_asynccnt, fragments via ds_load;
//    B streamed from L2; v_wmma_f32_16x16x32_bf16, fp32 accumulate.
//    Each wave owns 2 N-tiles x 4 M-tiles = 8 accumulators.
// ---------------------------------------------------------------------------

typedef __attribute__((ext_vector_type(16))) __bf16 v16bf;
typedef __attribute__((ext_vector_type(8)))  __bf16 v8bf;
typedef __attribute__((ext_vector_type(8)))  float  v8f;

#define B_  64
#define H_  512
#define V_  32000
#define T_  50
#define G_  2048     // 4*H
#define KA_ 1024     // E + H (fused gates GEMM K)

static __device__ __forceinline__ float sigf(float x) {
  return 1.0f / (1.0f + __expf(-x));
}

// A fragment (16x32 bf16, wave32): lane = half*16 + m.
// elements 0..7  = row m, K = kb + half*8  + [0..7]
// elements 8..15 = row m, K = kb + 16 + half*8 + [0..7]
static __device__ __forceinline__ v16bf load_a_frag(const __bf16* p) {
  v8bf lo = *(const v8bf*)(p);
  v8bf hi = *(const v8bf*)(p + 16);
  return __builtin_shufflevector(lo, hi, 0, 1, 2, 3, 4, 5, 6, 7,
                                 8, 9, 10, 11, 12, 13, 14, 15);
}

// Cooperative async stage of the 64 x KK bf16 A matrix into padded LDS.
// 16B chunks, one global_load_async_to_lds_b128 per chunk per lane.
template <int KK, int LW>
static __device__ __forceinline__ void stage_A_lds(
    const __bf16* __restrict__ g, __bf16* lds) {
  constexpr int CPR = KK / 8;            // 16B chunks per row
  constexpr int TOTAL = B_ * CPR;
  for (int cid = threadIdx.x; cid < TOTAL; cid += 128) {
    int row = cid / CPR;
    int col = (cid % CPR) * 8;
    const __bf16* src = g + (size_t)row * KK + col;
    // generic LDS address: low 32 bits == LDS byte offset
    unsigned int loff = (unsigned int)(uintptr_t)(lds + row * LW + col);
    asm volatile("global_load_async_to_lds_b128 %0, %1, off"
                 :: "v"(loff), "v"((unsigned long long)(uintptr_t)src)
                 : "memory");
  }
  asm volatile("s_wait_asynccnt 0" ::: "memory");
  __syncthreads();
}

#define WMMA_BF16(A, Bf, C) \
  __builtin_amdgcn_wmma_f32_16x16x32_bf16(false, (A), false, (Bf), (short)0, (C), false, false)

// ------------------------- one-time conversion kernels ---------------------

__global__ __launch_bounds__(256) void convert_wlin_kernel(
    const float* __restrict__ W, __bf16* __restrict__ Wb) {
  size_t i = ((size_t)blockIdx.x * 256 + threadIdx.x) * 8;  // V_*H_ total
#pragma unroll
  for (int e = 0; e < 8; ++e) Wb[i + e] = (__bf16)W[i + e];
}

__global__ __launch_bounds__(256) void build_wcat_kernel(
    const float* __restrict__ Wih, const float* __restrict__ Whh,
    __bf16* __restrict__ Wc) {
  size_t i = ((size_t)blockIdx.x * 256 + threadIdx.x) * 8;  // G_*KA_ total
#pragma unroll
  for (int e = 0; e < 8; ++e) {
    size_t idx = i + e;
    int r = (int)(idx >> 10);       // / KA_
    int k = (int)(idx & 1023);
    float v = (k < H_) ? Wih[(size_t)r * H_ + k] : Whh[(size_t)r * H_ + (k - H_)];
    Wc[idx] = (__bf16)v;
  }
}

__global__ __launch_bounds__(256) void init_state_kernel(
    const float* __restrict__ features, float* __restrict__ c,
    __bf16* __restrict__ hbf, __bf16* __restrict__ Agates,
    const int* __restrict__ start_tok, int* __restrict__ word,
    int* __restrict__ mask, int* __restrict__ done) {
  int b = blockIdx.x;
#pragma unroll
  for (int jj = 0; jj < 2; ++jj) {
    int j = threadIdx.x + jj * 256;
    float f = features[(size_t)b * H_ + j];
    __bf16 fb = (__bf16)f;
    hbf[(size_t)b * H_ + j] = fb;
    Agates[(size_t)b * KA_ + H_ + j] = fb;   // h half of [emb|h]
    c[(size_t)b * H_ + j] = 0.0f;
  }
  if (b == 0 && threadIdx.x < B_) {
    word[threadIdx.x] = *start_tok;
    mask[threadIdx.x] = 0;
    if (threadIdx.x == 0) *done = 0;
  }
}

// ------------------------------ per-step kernels ---------------------------

__global__ __launch_bounds__(256) void embed_kernel(
    const float* __restrict__ emb, const int* __restrict__ word,
    __bf16* __restrict__ Agates) {
  int b = blockIdx.x;
  int w = word[b];
#pragma unroll
  for (int jj = 0; jj < 2; ++jj) {
    int j = threadIdx.x + jj * 256;
    Agates[(size_t)b * KA_ + j] = (__bf16)emb[(size_t)w * H_ + j];
  }
}

// gates = A(64x1024) * Wcat^T(1024x2048) + (b_ih + b_hh)
// 16 blocks x 128 threads; wave = 2 N-tiles x 4 M-tiles.
__global__ __launch_bounds__(128) void gates_gemm_kernel(
    const __bf16* __restrict__ A, const __bf16* __restrict__ W,
    const float* __restrict__ b_ih, const float* __restrict__ b_hh,
    float* __restrict__ gates) {
  constexpr int LW = KA_ + 8;                  // padded LDS row (elems)
  __shared__ __bf16 Alds[B_ * LW];             // 132,096 B
  stage_A_lds<KA_, LW>(A, Alds);

  const int lane = threadIdx.x & 31;
  const int wave = threadIdx.x >> 5;
  const int n0   = (blockIdx.x * 8 + wave * 2) * 16;
  const int mr   = lane & 15;
  const int half = lane >> 4;

  const __bf16* brow0 = W + (size_t)(n0 + mr) * KA_ + half * 8;
  const __bf16* brow1 = brow0 + (size_t)16 * KA_;
  const __bf16* a0 = &Alds[(mr +  0) * LW + half * 8];
  const __bf16* a1 = &Alds[(mr + 16) * LW + half * 8];
  const __bf16* a2 = &Alds[(mr + 32) * LW + half * 8];
  const __bf16* a3 = &Alds[(mr + 48) * LW + half * 8];

  v8f acc00 = {}, acc10 = {}, acc20 = {}, acc30 = {};
  v8f acc01 = {}, acc11 = {}, acc21 = {}, acc31 = {};
  for (int kb = 0; kb < KA_; kb += 32) {
    v16bf bf0 = load_a_frag(brow0 + kb);
    v16bf bf1 = load_a_frag(brow1 + kb);
    v16bf af0 = load_a_frag(a0 + kb);
    v16bf af1 = load_a_frag(a1 + kb);
    v16bf af2 = load_a_frag(a2 + kb);
    v16bf af3 = load_a_frag(a3 + kb);
    acc00 = WMMA_BF16(af0, bf0, acc00);
    acc10 = WMMA_BF16(af1, bf0, acc10);
    acc20 = WMMA_BF16(af2, bf0, acc20);
    acc30 = WMMA_BF16(af3, bf0, acc30);
    acc01 = WMMA_BF16(af0, bf1, acc01);
    acc11 = WMMA_BF16(af1, bf1, acc11);
    acc21 = WMMA_BF16(af2, bf1, acc21);
    acc31 = WMMA_BF16(af3, bf1, acc31);
  }

  const int na = n0 + mr;                      // C/D column, tile 0
  const int nb = na + 16;                      // C/D column, tile 1
  const float biasa = b_ih[na] + b_hh[na];
  const float biasb = b_ih[nb] + b_hh[nb];
#pragma unroll
  for (int r = 0; r < 8; ++r) {
    int mrow = half * 8 + r;                   // C/D row layout
    gates[(size_t)(mrow +  0) * G_ + na] = acc00[r] + biasa;
    gates[(size_t)(mrow + 16) * G_ + na] = acc10[r] + biasa;
    gates[(size_t)(mrow + 32) * G_ + na] = acc20[r] + biasa;
    gates[(size_t)(mrow + 48) * G_ + na] = acc30[r] + biasa;
    gates[(size_t)(mrow +  0) * G_ + nb] = acc01[r] + biasb;
    gates[(size_t)(mrow + 16) * G_ + nb] = acc11[r] + biasb;
    gates[(size_t)(mrow + 32) * G_ + nb] = acc21[r] + biasb;
    gates[(size_t)(mrow + 48) * G_ + nb] = acc31[r] + biasb;
  }
}

__global__ __launch_bounds__(256) void lstm_cell_kernel(
    const float* __restrict__ gates, float* __restrict__ c,
    __bf16* __restrict__ hbf, __bf16* __restrict__ Agates) {
  int b = blockIdx.x;
#pragma unroll
  for (int jj = 0; jj < 2; ++jj) {
    int j = threadIdx.x + jj * 256;
    const float* g = gates + (size_t)b * G_;
    float ig = g[j];
    float fg = g[j + 512];
    float gg = g[j + 1024];
    float og = g[j + 1536];
    float cv = c[(size_t)b * H_ + j];
    float cn = sigf(fg) * cv + sigf(ig) * tanhf(gg);
    float hn = sigf(og) * tanhf(cn);
    c[(size_t)b * H_ + j] = cn;
    __bf16 hb = (__bf16)hn;
    hbf[(size_t)b * H_ + j] = hb;
    Agates[(size_t)b * KA_ + H_ + j] = hb;     // feed next step's gates GEMM
  }
}

// out[:, t, :] = done ? 0 : h(64x512) * Wlin^T(512x32000) + b_lin
// 250 blocks x 128 threads; wave = 2 N-tiles x 4 M-tiles.
__global__ __launch_bounds__(128) void logits_gemm_kernel(
    const __bf16* __restrict__ A, const __bf16* __restrict__ W,
    const float* __restrict__ blin, const int* __restrict__ done,
    float* __restrict__ out, int t) {
  constexpr int LW = H_ + 8;                   // padded LDS row (elems)
  __shared__ __bf16 Alds[B_ * LW];             // 66,560 B
  stage_A_lds<H_, LW>(A, Alds);

  const int lane = threadIdx.x & 31;
  const int wave = threadIdx.x >> 5;
  const int n0   = (blockIdx.x * 8 + wave * 2) * 16;
  const int mr   = lane & 15;
  const int half = lane >> 4;

  const __bf16* brow0 = W + (size_t)(n0 + mr) * H_ + half * 8;
  const __bf16* brow1 = brow0 + (size_t)16 * H_;
  const __bf16* a0 = &Alds[(mr +  0) * LW + half * 8];
  const __bf16* a1 = &Alds[(mr + 16) * LW + half * 8];
  const __bf16* a2 = &Alds[(mr + 32) * LW + half * 8];
  const __bf16* a3 = &Alds[(mr + 48) * LW + half * 8];

  v8f acc00 = {}, acc10 = {}, acc20 = {}, acc30 = {};
  v8f acc01 = {}, acc11 = {}, acc21 = {}, acc31 = {};
  for (int kb = 0; kb < H_; kb += 32) {
    v16bf bf0 = load_a_frag(brow0 + kb);
    v16bf bf1 = load_a_frag(brow1 + kb);
    v16bf af0 = load_a_frag(a0 + kb);
    v16bf af1 = load_a_frag(a1 + kb);
    v16bf af2 = load_a_frag(a2 + kb);
    v16bf af3 = load_a_frag(a3 + kb);
    acc00 = WMMA_BF16(af0, bf0, acc00);
    acc10 = WMMA_BF16(af1, bf0, acc10);
    acc20 = WMMA_BF16(af2, bf0, acc20);
    acc30 = WMMA_BF16(af3, bf0, acc30);
    acc01 = WMMA_BF16(af0, bf1, acc01);
    acc11 = WMMA_BF16(af1, bf1, acc11);
    acc21 = WMMA_BF16(af2, bf1, acc21);
    acc31 = WMMA_BF16(af3, bf1, acc31);
  }

  const int na = n0 + mr;
  const int nb = na + 16;
  const float biasa = blin[na];
  const float biasb = blin[nb];
  const int isdone = *done;
  const size_t rowstride = (size_t)T_ * V_;
  const size_t basea = (size_t)t * V_ + na;
  const size_t baseb = (size_t)t * V_ + nb;
#pragma unroll
  for (int r = 0; r < 8; ++r) {
    int mrow = half * 8 + r;
    out[(size_t)(mrow +  0) * rowstride + basea] = isdone ? 0.0f : (acc00[r] + biasa);
    out[(size_t)(mrow + 16) * rowstride + basea] = isdone ? 0.0f : (acc10[r] + biasa);
    out[(size_t)(mrow + 32) * rowstride + basea] = isdone ? 0.0f : (acc20[r] + biasa);
    out[(size_t)(mrow + 48) * rowstride + basea] = isdone ? 0.0f : (acc30[r] + biasa);
    out[(size_t)(mrow +  0) * rowstride + baseb] = isdone ? 0.0f : (acc01[r] + biasb);
    out[(size_t)(mrow + 16) * rowstride + baseb] = isdone ? 0.0f : (acc11[r] + biasb);
    out[(size_t)(mrow + 32) * rowstride + baseb] = isdone ? 0.0f : (acc21[r] + biasb);
    out[(size_t)(mrow + 48) * rowstride + baseb] = isdone ? 0.0f : (acc31[r] + biasb);
  }
}

// first-occurrence argmax over V per batch row; update word/mask
__global__ __launch_bounds__(256) void argmax_kernel(
    const float* __restrict__ out, int t, const int* __restrict__ end_tok,
    int* __restrict__ word, int* __restrict__ mask) {
  __shared__ float sv[256];
  __shared__ int   si[256];
  int b = blockIdx.x;
  const float* row = out + ((size_t)b * T_ + t) * V_;
  float best = -__builtin_inff();
  int   bi   = 0x7fffffff;
  for (int v = threadIdx.x; v < V_; v += 256) {
    float x = row[v];
    if (x > best) { best = x; bi = v; }       // ascending scan: first max kept
  }
  sv[threadIdx.x] = best;
  si[threadIdx.x] = bi;
  __syncthreads();
  for (int s = 128; s > 0; s >>= 1) {
    if (threadIdx.x < (unsigned)s) {
      float ov = sv[threadIdx.x + s];
      int   oi = si[threadIdx.x + s];
      if (ov > sv[threadIdx.x] ||
          (ov == sv[threadIdx.x] && oi < si[threadIdx.x])) {
        sv[threadIdx.x] = ov;
        si[threadIdx.x] = oi;
      }
    }
    __syncthreads();
  }
  if (threadIdx.x == 0) {
    int w = si[0];
    word[b] = w;
    if (w == *end_tok) mask[b] = 1;
  }
}

__global__ __launch_bounds__(64) void done_kernel(
    const int* __restrict__ mask, int* __restrict__ done) {
  __shared__ int s[64];
  s[threadIdx.x] = mask[threadIdx.x];
  __syncthreads();
  for (int st = 32; st > 0; st >>= 1) {
    if (threadIdx.x < (unsigned)st) s[threadIdx.x] &= s[threadIdx.x + st];
    __syncthreads();
  }
  if (threadIdx.x == 0 && s[0]) *done = 1;
}

// ------------------------------ host wiring --------------------------------

extern "C" void kernel_launch(void* const* d_in, const int* in_sizes, int n_in,
                              void* d_out, int out_size, void* d_ws, size_t ws_size,
                              hipStream_t stream) {
  (void)in_sizes; (void)n_in; (void)out_size; (void)ws_size;

  const float* features  = (const float*)d_in[0];
  const float* embedding = (const float*)d_in[1];
  const float* W_ih      = (const float*)d_in[2];
  const float* W_hh      = (const float*)d_in[3];
  const float* b_ih      = (const float*)d_in[4];
  const float* b_hh      = (const float*)d_in[5];
  const float* W_lin     = (const float*)d_in[6];
  const float* b_lin     = (const float*)d_in[7];
  const int*   start_tok = (const int*)d_in[8];
  const int*   end_tok   = (const int*)d_in[9];
  float*       out       = (float*)d_out;

  // workspace layout (all blocks 256B-aligned)
  char* ws = (char*)d_ws;
  constexpr size_t SZ_WLIN = (size_t)V_ * H_ * 2;   // bf16 W_lin
  constexpr size_t SZ_WCAT = (size_t)G_ * KA_ * 2;  // bf16 [W_ih|W_hh]
  constexpr size_t SZ_AG   = (size_t)B_ * KA_ * 2;  // bf16 [emb|h]
  constexpr size_t SZ_HBF  = (size_t)B_ * H_ * 2;   // bf16 h
  constexpr size_t SZ_C    = (size_t)B_ * H_ * 4;   // fp32 c
  constexpr size_t SZ_G    = (size_t)B_ * G_ * 4;   // fp32 gates

  __bf16* Wlin_bf = (__bf16*)ws;                 ws += SZ_WLIN;
  __bf16* Wcat_bf = (__bf16*)ws;                 ws += SZ_WCAT;
  __bf16* Agates  = (__bf16*)ws;                 ws += SZ_AG;
  __bf16* hbf     = (__bf16*)ws;                 ws += SZ_HBF;
  float*  c       = (float*)ws;                  ws += SZ_C;
  float*  gates   = (float*)ws;                  ws += SZ_G;
  int*    word    = (int*)ws;                    ws += 256;
  int*    mask    = (int*)ws;                    ws += 256;
  int*    done    = (int*)ws;                    ws += 256;

  convert_wlin_kernel<<<(V_ * H_) / (256 * 8), 256, 0, stream>>>(W_lin, Wlin_bf);
  build_wcat_kernel<<<(G_ * KA_) / (256 * 8), 256, 0, stream>>>(W_ih, W_hh, Wcat_bf);
  init_state_kernel<<<B_, 256, 0, stream>>>(features, c, hbf, Agates,
                                            start_tok, word, mask, done);

  for (int t = 0; t < T_; ++t) {
    embed_kernel<<<B_, 256, 0, stream>>>(embedding, word, Agates);
    gates_gemm_kernel<<<G_ / 128, 128, 0, stream>>>(Agates, Wcat_bf, b_ih, b_hh, gates);
    lstm_cell_kernel<<<B_, 256, 0, stream>>>(gates, c, hbf, Agates);
    logits_gemm_kernel<<<V_ / 128, 128, 0, stream>>>(hbf, Wlin_bf, b_lin, done, out, t);
    argmax_kernel<<<B_, 256, 0, stream>>>(out, t, end_tok, word, mask);
    done_kernel<<<1, 64, 0, stream>>>(mask, done);
  }
}
